// Deformable_Attention_88459146428622
// MI455X (gfx1250) — matrix operand order repaired and checked by reading, not compiled, not run on hardware
//
#include <hip/hip_runtime.h>
#include <hip/hip_bf16.h>

#define T_LEN 2048
#define E_DIM 512
#define NHEAD 8
#define HD_DIM 64
#define SCALE 0.125f

typedef __attribute__((ext_vector_type(16))) __bf16 v16bf;
typedef __attribute__((ext_vector_type(8)))  float  v8f;

union BV {
    v16bf v;
    unsigned short u[16];
    uint4 q[2];
};

__device__ __forceinline__ unsigned short f2bf(float f) {
    unsigned u = __builtin_bit_cast(unsigned, f);
    u += 0x7FFFu + ((u >> 16) & 1u);   // round-to-nearest-even
    return (unsigned short)(u >> 16);
}

__device__ __forceinline__ unsigned pack2(float lo, float hi) {
    return (unsigned)f2bf(lo) | ((unsigned)f2bf(hi) << 16);
}

__device__ __forceinline__ v8f wmma_bf16(v16bf a, v16bf b, v8f c) {
    return __builtin_amdgcn_wmma_f32_16x16x32_bf16(false, a, false, b, (short)0, c, false, false);
}

// ---------------------------------------------------------------------------
// Prep A: x (f32 row-major) -> xbf (bf16 row-major). 8 elements / thread.
// ---------------------------------------------------------------------------
__global__ __launch_bounds__(256) void k_cvt_x(const float* __restrict__ x,
                                               unsigned short* __restrict__ xbf) {
    const int i = (blockIdx.x * 256 + threadIdx.x) * 8;
    float4 f0 = *(const float4*)(x + i);
    float4 f1 = *(const float4*)(x + i + 4);
    uint4 p;
    p.x = pack2(f0.x, f0.y);
    p.y = pack2(f0.z, f0.w);
    p.z = pack2(f1.x, f1.y);
    p.w = pack2(f1.z, f1.w);
    *(uint4*)(xbf + i) = p;
}

// ---------------------------------------------------------------------------
// Prep B: w (f32, K x N row-major) -> wT (bf16, N x K row-major).
// One element per thread; writes coalesced along k.
// ---------------------------------------------------------------------------
__global__ __launch_bounds__(256) void k_cvt_wT(const float* __restrict__ w,
                                                unsigned short* __restrict__ wT,
                                                int K, int N) {
    const int id = blockIdx.x * 256 + threadIdx.x;   // id = n*K + k
    const int n = id / K, k = id % K;
    wT[id] = f2bf(w[k * N + n]);
}

// ---------------------------------------------------------------------------
// Kernel 1: qkv = x @ w_qkv + b_qkv.  One wave per 16x16 tile, K=512 (16 steps).
// A from xbf (bf16 row-major), B from wqkvT (bf16 N-major) -> pure b128 loads.
// Scatter: Q -> Qf (f32) + Qbf (bh,t,hd) ; K -> Kbf (bh,t,hd) ; V -> VbfT (bh,hd,t)
// ---------------------------------------------------------------------------
__global__ __launch_bounds__(256) void k_qkv(const unsigned short* __restrict__ xbf,
                                             const unsigned short* __restrict__ wT,
                                             const float* __restrict__ bias,
                                             float* __restrict__ Qf,
                                             unsigned short* __restrict__ Qbf,
                                             unsigned short* __restrict__ Kbf,
                                             unsigned short* __restrict__ VbfT) {
    const int lane = threadIdx.x & 31;
    const int wv   = threadIdx.x >> 5;
    const int tile = blockIdx.x * 8 + wv;          // 24576 tiles = 256 x 96
    const int tm = tile / 96, tn = tile % 96;
    const int rbase = tm * 16, nbase = tn * 16;
    const int ln = lane & 15, hi = lane >> 4;
    const int ncol = nbase + ln;

    v8f c = {};
    const unsigned short* arow = xbf + (rbase + ln) * E_DIM;
    const unsigned short* brow = wT + ncol * E_DIM;
#pragma unroll 4
    for (int k0 = 0; k0 < E_DIM; k0 += 32) {
        BV a, bm;
        a.q[0]  = *(const uint4*)(arow + k0 + 16 * hi);
        a.q[1]  = *(const uint4*)(arow + k0 + 16 * hi + 8);
        bm.q[0] = *(const uint4*)(brow + k0 + 16 * hi);
        bm.q[1] = *(const uint4*)(brow + k0 + 16 * hi + 8);
        c = wmma_bf16(a.v, bm.v, c);
    }

    const float bvl = bias[ncol];
    const int mat = ncol / E_DIM;          // uniform across the wave (tile of 16 cols)
    const int e   = ncol % E_DIM;
    const int h   = e >> 6, hd = e & 63;
#pragma unroll
    for (int v = 0; v < 8; v++) {
        const int r = rbase + v + 8 * hi;
        const int b = r >> 11, t = r & (T_LEN - 1);
        const float val = c[v] + bvl;
        const unsigned short bf = f2bf(val);
        const int bh = b * NHEAD + h;
        if (mat == 0) {
            Qf[r * E_DIM + e] = val;
            Qbf[(bh * T_LEN + t) * HD_DIM + hd] = bf;
        } else if (mat == 1) {
            Kbf[(bh * T_LEN + t) * HD_DIM + hd] = bf;
        } else {
            VbfT[(bh * HD_DIM + hd) * T_LEN + t] = bf;
        }
    }
}

// ---------------------------------------------------------------------------
// Kernel 2: od = Q @ w_od + b_od -> window params {start, end, anchor} per (bh,t)
// ---------------------------------------------------------------------------
__global__ __launch_bounds__(256) void k_od(const float* __restrict__ Qf,
                                            const float* __restrict__ w_od,
                                            const float* __restrict__ b_od,
                                            float4* __restrict__ params) {
    __shared__ float sod[16][17];
    const int r16 = threadIdx.x >> 4, col = threadIdx.x & 15;
    const int row = blockIdx.x * 16 + r16;
    float acc = b_od[col];
    const float* q = Qf + row * E_DIM;
    for (int k = 0; k < E_DIM; k++) acc += q[k] * w_od[k * 16 + col];
    sod[r16][col] = acc;
    __syncthreads();
    if (col < NHEAD) {
        const int b = row >> 11, t = row & (T_LEN - 1);
        const float off = tanhf(sod[r16][col]) * (float)T_LEN;
        const float dur = (1.0f / (1.0f + expf(-sod[r16][col + NHEAD]))) * (float)T_LEN;
        const float anchor = (float)t + off;
        params[(b * NHEAD + col) * T_LEN + t] =
            make_float4(anchor - dur, anchor + dur, anchor, 0.0f);
    }
}

// ---------------------------------------------------------------------------
// Kernel 3: fused flash attention. One wave = (head, 16-query tile).
// Loops over 64 key blocks of 32 keys: S via 4 bf16 WMMAs, point-weight +
// window mask, online softmax (shfl_xor row reductions), P transposed through
// LDS, O accumulated via 4 bf16 WMMAs with rescale. Output stored bf16.
// ---------------------------------------------------------------------------
__global__ __launch_bounds__(256) void k_attn(const unsigned short* __restrict__ Qbf,
                                              const unsigned short* __restrict__ Kbf,
                                              const unsigned short* __restrict__ VbfT,
                                              const float4* __restrict__ params,
                                              unsigned short* __restrict__ Aout) {
    __shared__ alignas(16) unsigned short sP[8][16 * 32];
    const int lane = threadIdx.x & 31;
    const int wv   = threadIdx.x >> 5;
    const int wid  = blockIdx.x * 8 + wv;        // 2048 waves = 16 heads * 128 qtiles
    const int bh = wid >> 7;
    const int tq = (wid & 127) * 16;
    const int ln = lane & 15, hi = lane >> 4;

    // per-lane query-row window params (rows duplicated on both lane halves)
    const float4 pp = params[bh * T_LEN + tq + ln];
    const float p_start = pp.x, p_end = pp.y, p_anchor = pp.z;

    // Q tile in A layout, K-dim slices 0..31 and 32..63
    const unsigned short* qrow = Qbf + (bh * T_LEN + tq + ln) * HD_DIM;
    BV a0, a1;
    a0.q[0] = *(const uint4*)(qrow + 16 * hi);
    a0.q[1] = *(const uint4*)(qrow + 16 * hi + 8);
    a1.q[0] = *(const uint4*)(qrow + 32 + 16 * hi);
    a1.q[1] = *(const uint4*)(qrow + 32 + 16 * hi + 8);

    v8f O[4];
#pragma unroll
    for (int n = 0; n < 4; n++) O[n] = (v8f){};
    float M[8], L[8], fs[8];
#pragma unroll
    for (int v = 0; v < 8; v++) { M[v] = -3.0e38f; L[v] = 0.0f; }

    unsigned short* myP = &sP[wv][0];

    for (int kb = 0; kb < T_LEN / 32; kb++) {
        const int jbase = kb * 32;
        // ---- S = Q K^T for two 16-key column groups
        v8f s0 = {}, s1 = {};
        {
            const unsigned short* kr0 = Kbf + (bh * T_LEN + jbase + ln) * HD_DIM;
            const unsigned short* kr1 = Kbf + (bh * T_LEN + jbase + 16 + ln) * HD_DIM;
            BV b00, b01, b10, b11;
            b00.q[0] = *(const uint4*)(kr0 + 16 * hi);      b00.q[1] = *(const uint4*)(kr0 + 16 * hi + 8);
            b01.q[0] = *(const uint4*)(kr0 + 32 + 16 * hi); b01.q[1] = *(const uint4*)(kr0 + 32 + 16 * hi + 8);
            b10.q[0] = *(const uint4*)(kr1 + 16 * hi);      b10.q[1] = *(const uint4*)(kr1 + 16 * hi + 8);
            b11.q[0] = *(const uint4*)(kr1 + 32 + 16 * hi); b11.q[1] = *(const uint4*)(kr1 + 32 + 16 * hi + 8);
            s0 = wmma_bf16(a0.v, b00.v, s0);
            s0 = wmma_bf16(a1.v, b01.v, s0);
            s1 = wmma_bf16(a0.v, b10.v, s1);
            s1 = wmma_bf16(a1.v, b11.v, s1);
        }

        const float j0 = (float)(jbase + ln);
        const float j1 = (float)(jbase + 16 + ln);
        // ---- scores -> point weight -> mask -> online softmax, write P to LDS
#pragma unroll
        for (int v = 0; v < 8; v++) {
            const int src = v + 24 * hi;            // lane holding params of row v+8*hi
            const float st = __shfl(p_start,  src, 32);
            const float en = __shfl(p_end,    src, 32);
            const float an = __shfl(p_anchor, src, 32);
            const float bl = floorf(st), br = ceilf(en);
            const float al = floorf(an), ar = al + 1.0f, adl = an - al;

            float pw0 = 1.0f + (j0 == bl ? (bl - st) : 0.0f) + (j0 == br ? (en - br) : 0.0f)
                             + (j0 == ar ? adl : 0.0f)       + (j0 == al ? (1.0f - adl) : 0.0f);
            float pw1 = 1.0f + (j1 == bl ? (bl - st) : 0.0f) + (j1 == br ? (en - br) : 0.0f)
                             + (j1 == ar ? adl : 0.0f)       + (j1 == al ? (1.0f - adl) : 0.0f);
            float sc0 = s0[v] * SCALE * pw0;
            float sc1 = s1[v] * SCALE * pw1;
            if (j0 < bl || j0 > br) sc0 = -1.0e8f;
            if (j1 < bl || j1 > br) sc1 = -1.0e8f;

            // row max over the 16 lanes of this row (xor 1,2,4,8 stays in the half)
            float lm = fmaxf(sc0, sc1);
            lm = fmaxf(lm, __shfl_xor(lm, 1, 32));
            lm = fmaxf(lm, __shfl_xor(lm, 2, 32));
            lm = fmaxf(lm, __shfl_xor(lm, 4, 32));
            lm = fmaxf(lm, __shfl_xor(lm, 8, 32));
            const float mn = fmaxf(M[v], lm);
            const float f  = expf(M[v] - mn);
            M[v] = mn;
            const float p0 = expf(sc0 - mn);
            const float p1 = expf(sc1 - mn);
            float rs = p0 + p1;
            rs += __shfl_xor(rs, 1, 32);
            rs += __shfl_xor(rs, 2, 32);
            rs += __shfl_xor(rs, 4, 32);
            rs += __shfl_xor(rs, 8, 32);
            L[v] = L[v] * f + rs;
            fs[v] = f;
            myP[(v + 8 * hi) * 32 + ln]      = f2bf(p0);
            myP[(v + 8 * hi) * 32 + 16 + ln] = f2bf(p1);
        }

        // ensure the cross-lane LDS transpose is visible before reload
        asm volatile("s_wait_dscnt 0" ::: "memory");

        BV pa;   // P in A layout (16 queries x 32 keys)
    pa.q[0] = *(const uint4*)(myP + ln * 32 + 16 * hi);
    pa.q[1] = *(const uint4*)(myP + ln * 32 + 16 * hi + 8);

        // ---- rescale accumulators, then O += P * V
#pragma unroll
        for (int n = 0; n < 4; n++)
#pragma unroll
            for (int v = 0; v < 8; v++) O[n][v] *= fs[v];

#pragma unroll
        for (int n = 0; n < 4; n++) {
            const unsigned short* vr =
                VbfT + (bh * HD_DIM + n * 16 + ln) * T_LEN + jbase + 16 * hi;
            BV bv;
            bv.q[0] = *(const uint4*)(vr);
            bv.q[1] = *(const uint4*)(vr + 8);
            O[n] = wmma_bf16(pa.v, bv.v, O[n]);
        }
    }

    // ---- finalize: divide by row sums, scatter to (B,T,E) bf16 layout
    const int h = bh & (NHEAD - 1), b = bh >> 3;
#pragma unroll
    for (int v = 0; v < 8; v++) {
        const float inv = 1.0f / L[v];
        const int row = b * T_LEN + tq + v + 8 * hi;
#pragma unroll
        for (int n = 0; n < 4; n++)
            Aout[row * E_DIM + h * HD_DIM + n * 16 + ln] = f2bf(O[n][v] * inv);
    }
}

// ---------------------------------------------------------------------------
// Kernel 4: out = Aout @ w_out + b_out  (4096x512 @ 512x512), f32 result.
// A bf16 row-major, B = w_outT bf16 N-major -> pure b128 loads.
// ---------------------------------------------------------------------------
__global__ __launch_bounds__(256) void k_out(const unsigned short* __restrict__ A,
                                             const unsigned short* __restrict__ wT,
                                             const float* __restrict__ bias,
                                             float* __restrict__ out) {
    const int lane = threadIdx.x & 31;
    const int wv   = threadIdx.x >> 5;
    const int tile = blockIdx.x * 8 + wv;          // 8192 tiles = 256 x 32
    const int tm = tile / 32, tn = tile % 32;
    const int rbase = tm * 16, nbase = tn * 16;
    const int ln = lane & 15, hi = lane >> 4;
    const int ncol = nbase + ln;

    v8f c = {};
    const unsigned short* arow = A + (rbase + ln) * E_DIM;
    const unsigned short* brow = wT + ncol * E_DIM;
#pragma unroll 4
    for (int k0 = 0; k0 < E_DIM; k0 += 32) {
        BV a, bm;
        a.q[0]  = *(const uint4*)(arow + k0 + 16 * hi);
        a.q[1]  = *(const uint4*)(arow + k0 + 16 * hi + 8);
        bm.q[0] = *(const uint4*)(brow + k0 + 16 * hi);
        bm.q[1] = *(const uint4*)(brow + k0 + 16 * hi + 8);
        c = wmma_bf16(a.v, bm.v, c);
    }
    const float bvl = bias[ncol];
#pragma unroll
    for (int v = 0; v < 8; v++)
        out[(rbase + v + 8 * hi) * E_DIM + ncol] = c[v] + bvl;
}

// ---------------------------------------------------------------------------
extern "C" void kernel_launch(void* const* d_in, const int* in_sizes, int n_in,
                              void* d_out, int out_size, void* d_ws, size_t ws_size,
                              hipStream_t stream) {
    (void)in_sizes; (void)n_in; (void)out_size; (void)ws_size;
    const float* x     = (const float*)d_in[0];
    const float* w_qkv = (const float*)d_in[1];
    const float* b_qkv = (const float*)d_in[2];
    const float* w_od  = (const float*)d_in[3];
    const float* b_od  = (const float*)d_in[4];
    const float* w_out = (const float*)d_in[5];
    const float* b_out = (const float*)d_in[6];
    float* out = (float*)d_out;

    char* ws = (char*)d_ws;
    size_t o = 0;
    float*          Qf     = (float*)(ws + o);          o += (size_t)4096 * 512 * 4;
    unsigned short* Qbf    = (unsigned short*)(ws + o); o += (size_t)4096 * 512 * 2;
    unsigned short* Kbf    = (unsigned short*)(ws + o); o += (size_t)4096 * 512 * 2;
    unsigned short* VbfT   = (unsigned short*)(ws + o); o += (size_t)4096 * 512 * 2;
    float4*         params = (float4*)(ws + o);         o += (size_t)16 * 2048 * 16;
    unsigned short* AoutBf = (unsigned short*)(ws + o); o += (size_t)4096 * 512 * 2;
    unsigned short* xbf    = (unsigned short*)(ws + o); o += (size_t)4096 * 512 * 2;
    unsigned short* wqkvT  = (unsigned short*)(ws + o); o += (size_t)1536 * 512 * 2;
    unsigned short* woutT  = (unsigned short*)(ws + o); o += (size_t)512 * 512 * 2;

    k_cvt_x <<<1024, 256, 0, stream>>>(x, xbf);                       // 2M elems, 8/thread
    k_cvt_wT<<<3072, 256, 0, stream>>>(w_qkv, wqkvT, E_DIM, 3 * E_DIM);
    k_cvt_wT<<<1024, 256, 0, stream>>>(w_out, woutT, E_DIM, E_DIM);

    k_qkv <<<3072, 256, 0, stream>>>(xbf, wqkvT, b_qkv, Qf, Qbf, Kbf, VbfT);
    k_od  <<<256,  256, 0, stream>>>(Qf, w_od, b_od, params);
    k_attn<<<256,  256, 0, stream>>>(Qbf, Kbf, VbfT, params, AoutBf);
    k_out <<<1024, 256, 0, stream>>>(AoutBf, woutT, b_out, out);
}
